// BasicLayer_9122510537170
// MI455X (gfx1250) — matrix operand order repaired and checked by reading, not compiled, not run on hardware
//
#include <hip/hip_runtime.h>

typedef __bf16 bf16_t;
typedef bf16_t v16bf __attribute__((ext_vector_type(16)));
typedef bf16_t v8bf  __attribute__((ext_vector_type(8)));
typedef float  v8f   __attribute__((ext_vector_type(8)));
typedef unsigned int v4u __attribute__((ext_vector_type(4)));
typedef int v4i __attribute__((ext_vector_type(4)));
typedef int v8i __attribute__((ext_vector_type(8)));

#define S_DIM 32
#define H_DIM 64
#define W_DIM 64
#define C_DIM 192
#define NH    6
#define HD    32
#define NTOK  64
#define NWIN  2048
#define L_TOK 131072
#define C3    576
#define C4    768

#if defined(__gfx1250__) && __has_builtin(__builtin_amdgcn_tensor_load_to_lds) && __has_builtin(__builtin_amdgcn_s_wait_tensorcnt)
#define HAVE_TDM 1
#else
#define HAVE_TDM 0
#endif

static __device__ inline v16bf cat8(v8bf lo, v8bf hi) {
  return __builtin_shufflevector(lo, hi, 0,1,2,3,4,5,6,7,8,9,10,11,12,13,14,15);
}

// A fragment, 16x32 bf16, A row-major with leading dim ldA.
// lane<16: row=rowTile+lane, K = {k0..k0+7, k0+16..k0+23}
// lane>=16: row=rowTile+lane-16, K = {k0+8..k0+15, k0+24..k0+31}
static __device__ inline v16bf load_frag_A(const bf16_t* A, int ldA, int rowTile, int k0, int lane) {
  int l16 = lane & 15, hh = lane >> 4;
  const bf16_t* p = A + (size_t)(rowTile + l16) * ldA + k0 + hh * 8;
  v8bf lo = *(const v8bf*)p;
  v8bf hi = *(const v8bf*)(p + 16);
  return cat8(lo, hi);
}

// B fragment, 32x16 bf16, from W^T stored (N x K) row-major, leading dim ldB.
// lane<16: col=colTile+lane, K = {k0..k0+7, k0+16..k0+23}; lane>=16: K = {k0+8..k0+15, k0+24..k0+31}
static __device__ inline v16bf load_frag_B(const bf16_t* WT, int ldB, int colTile, int k0, int lane) {
  int l16 = lane & 15, hh = lane >> 4;
  const bf16_t* p = WT + (size_t)(colTile + l16) * ldB + k0 + hh * 8;
  v8bf lo = *(const v8bf*)p;
  v8bf hi = *(const v8bf*)(p + 16);
  return cat8(lo, hi);
}

static __device__ inline v8f wmma_bf16(v16bf a, v16bf b, v8f c) {
  return __builtin_amdgcn_wmma_f32_16x16x32_bf16(false, a, false, b, (short)0, c, false, false);
}

// ---------------------------------------------------------------- utilities

__global__ void copy_f32(const float* __restrict__ in, float* __restrict__ out, int n4) {
  const float4* i4 = (const float4*)in;
  float4* o4 = (float4*)out;
  for (int i = blockIdx.x * blockDim.x + threadIdx.x; i < n4; i += gridDim.x * blockDim.x)
    o4[i] = i4[i];
}

// in: (K x N) fp32 row-major  ->  out: (N x K) bf16 row-major (transposed)
__global__ void cvt_transpose(const float* __restrict__ in, bf16_t* __restrict__ out, int K, int N) {
  int total = K * N;
  for (int i = blockIdx.x * blockDim.x + threadIdx.x; i < total; i += gridDim.x * blockDim.x) {
    int k = i / N, n = i - k * N;
    out[(size_t)n * K + k] = (bf16_t)in[i];
  }
}

// ------------------------------------------------- LN + shift + window part
__global__ __launch_bounds__(256) void ln_part(const float* __restrict__ x,
                                               const float* __restrict__ g,
                                               const float* __restrict__ b,
                                               bf16_t* __restrict__ hout, int shifted) {
  int wave = threadIdx.x >> 5, lane = threadIdx.x & 31;
  int token = blockIdx.x * 8 + wave;
  int wc = token & 63, hc = (token >> 6) & 63, sc = token >> 12;

  const float* row = x + (size_t)token * C_DIM;
  float v[6], s = 0.f;
#pragma unroll
  for (int i = 0; i < 6; ++i) { v[i] = row[lane + 32 * i]; s += v[i]; }
#pragma unroll
  for (int off = 16; off > 0; off >>= 1) s += __shfl_xor(s, off, 32);
  float mean = s * (1.f / C_DIM);
  float var = 0.f;
#pragma unroll
  for (int i = 0; i < 6; ++i) { float d = v[i] - mean; var += d * d; }
#pragma unroll
  for (int off = 16; off > 0; off >>= 1) var += __shfl_xor(var, off, 32);
  float rstd = rsqrtf(var * (1.f / C_DIM) + 1e-5f);

  int ss = sc, hh = hc, ww = wc;
  if (shifted) { ss = (sc + S_DIM - 2) & (S_DIM - 1); hh = (hc + H_DIM - 2) & (H_DIM - 1); ww = (wc + W_DIM - 2) & (W_DIM - 1); }
  int win = (ss >> 2) * 256 + (hh >> 2) * 16 + (ww >> 2);
  int n   = (ss & 3) * 16 + (hh & 3) * 4 + (ww & 3);
  bf16_t* drow = hout + ((size_t)win * NTOK + n) * C_DIM;
#pragma unroll
  for (int i = 0; i < 6; ++i) {
    int c = lane + 32 * i;
    drow[c] = (bf16_t)((v[i] - mean) * rstd * g[c] + b[c]);
  }
}

// ------------------------------------------------------------- QKV GEMM
// out(L x 576) = A(L x 192) @ W + bias.  grid(2048, 9), 128 thr (4 waves),
// each wave computes a 32x32 register tile (2x2 WMMA tiles): 4 WMMA / 4 frag loads.
__global__ __launch_bounds__(128) void gemm_qkv(const bf16_t* __restrict__ A,
                                                const bf16_t* __restrict__ WT,
                                                const float* __restrict__ bias,
                                                bf16_t* __restrict__ out) {
  int wave = threadIdx.x >> 5, lane = threadIdx.x & 31;
  int row0 = blockIdx.x * 64 + (wave >> 1) * 32;
  int col0 = blockIdx.y * 64 + (wave & 1) * 32;
  v8f acc00 = {0,0,0,0,0,0,0,0}, acc01 = {0,0,0,0,0,0,0,0};
  v8f acc10 = {0,0,0,0,0,0,0,0}, acc11 = {0,0,0,0,0,0,0,0};
#pragma unroll
  for (int k0 = 0; k0 < C_DIM; k0 += 32) {
    v16bf a0 = load_frag_A(A, C_DIM, row0,      k0, lane);
    v16bf a1 = load_frag_A(A, C_DIM, row0 + 16, k0, lane);
    v16bf b0 = load_frag_B(WT, C_DIM, col0,      k0, lane);
    v16bf b1 = load_frag_B(WT, C_DIM, col0 + 16, k0, lane);
    acc00 = wmma_bf16(a0, b0, acc00);
    acc01 = wmma_bf16(a0, b1, acc01);
    acc10 = wmma_bf16(a1, b0, acc10);
    acc11 = wmma_bf16(a1, b1, acc11);
  }
  int l16 = lane & 15, hh = lane >> 4;
  float bb0 = bias[col0 + l16], bb1 = bias[col0 + 16 + l16];
#pragma unroll
  for (int j = 0; j < 8; ++j) {
    int ra = row0 + j + hh * 8, rb = ra + 16;
    out[(size_t)ra * C3 + col0 + l16]      = (bf16_t)(acc00[j] + bb0);
    out[(size_t)ra * C3 + col0 + 16 + l16] = (bf16_t)(acc01[j] + bb1);
    out[(size_t)rb * C3 + col0 + l16]      = (bf16_t)(acc10[j] + bb0);
    out[(size_t)rb * C3 + col0 + 16 + l16] = (bf16_t)(acc11[j] + bb1);
  }
}

// -------------------------------------------------------------- attention
static __device__ inline int region3(int c, int D) { return c < D - 4 ? 0 : (c < D - 2 ? 1 : 2); }

struct AttnSmem {            // rpbS must stay first (LDS byte offset 0 for TDM)
  float  rpbS[343 * NH];
  int    rcS[NTOK];
  bf16_t vT[HD][NTOK];
  bf16_t pmat[NTOK][NTOK];
};

// grid(6, 2048), 128 threads (4 waves). obuf bf16 windowed (L x 192).
__global__ __launch_bounds__(128) void attn_kernel(const bf16_t* __restrict__ qkv,
                                                   const float* __restrict__ rpb,
                                                   bf16_t* __restrict__ obuf, int shifted) {
  __shared__ __align__(16) AttnSmem sm;

  int head = blockIdx.x, win = blockIdx.y;
  int tid = threadIdx.x, wave = tid >> 5, lane = tid & 31;
  int l16 = lane & 15, hh = lane >> 4;

#if HAVE_TDM
  // Tensor Data Mover: DMA the 343x6 fp32 bias table into LDS (2-group D#,
  // VADDR2/VADDR3 zero-filled -> <=2D tensor).
  if (wave == 0) {
    unsigned long long ga = (unsigned long long)(const void*)rpb;
    const unsigned td0 = 343u * NH;            // elements (fp32)
    v4u g0; v8i g1; v4i z4 = {0, 0, 0, 0}; v8i z8 = {0, 0, 0, 0, 0, 0, 0, 0};
    g0[0] = 1u;                                // count=1, user descriptor
    g0[1] = 0u;                                // lds_addr = offset of rpbS = 0
    g0[2] = (unsigned)(ga & 0xffffffffu);      // global_addr[31:0]
    g0[3] = (unsigned)((ga >> 32) & 0x1ffffffu) | (2u << 30);  // addr[56:32] | type=2
    g1[0] = (int)(2u << 16);                   // wg_mask=0, data_size=2 (4B)
    g1[1] = (int)((td0 & 0xffffu) << 16);      // tensor_dim0[15:0]
    g1[2] = (int)((td0 >> 16) | (1u << 16));   // tensor_dim0[31:16], tensor_dim1=1
    g1[3] = (int)((td0 & 0xffffu) << 16);      // tile_dim0 = td0
    g1[4] = 1;                                 // tile_dim1 = 1, tile_dim2 = 0
    g1[5] = (int)td0;                          // tensor_dim0_stride[31:0]
    g1[6] = (int)((td0 & 0xffffu) << 16);      // stride0[47:32]=0 | stride1[15:0]
    g1[7] = 0;                                 // stride1[47:16] = 0
    __builtin_amdgcn_tensor_load_to_lds(g0, g1, z4, z4, z8, 0);
    __builtin_amdgcn_s_wait_tensorcnt((short)0);
  }
#else
  for (int i = tid; i < 343 * NH; i += 128) sm.rpbS[i] = rpb[i];
#endif
  if (tid < NTOK) {
    int rc = 0;
    if (shifted) {
      int s4 = win >> 8, h4 = (win >> 4) & 15, w4 = win & 15;
      int scc = s4 * 4 + (tid >> 4), hcc = h4 * 4 + ((tid >> 2) & 3), wcc = w4 * 4 + (tid & 3);
      rc = region3(scc, S_DIM) * 9 + region3(hcc, H_DIM) * 3 + region3(wcc, W_DIM);
    }
    sm.rcS[tid] = rc;
  }
  const bf16_t* vbase = qkv + (size_t)win * NTOK * C3 + 2 * C_DIM + head * HD;
  for (int i = tid; i < NTOK * HD; i += 128) {
    int tok = i >> 5, d = i & 31;
    sm.vT[d][tok] = vbase[(size_t)tok * C3 + d];
  }
  __syncthreads();

  // logits: wave handles M rows [wave*16, wave*16+16), all 4 N tiles (K = HD = 32)
  const bf16_t* qbase = qkv + (size_t)win * NTOK * C3 + head * HD;
  const bf16_t* kbase = qkv + (size_t)win * NTOK * C3 + C_DIM + head * HD;
  v8f acc[4];
#pragma unroll
  for (int t = 0; t < 4; ++t) acc[t] = (v8f){0,0,0,0,0,0,0,0};
  {
    v16bf a = load_frag_A(qbase, C3, wave * 16, 0, lane);
#pragma unroll
    for (int t = 0; t < 4; ++t) {
      v16bf b = load_frag_B(kbase, C3, t * 16, 0, lane);
      acc[t] = wmma_bf16(a, b, acc[t]);
    }
  }
  const float scale = 0.17677669529663687f; // 1/sqrt(32)
#pragma unroll
  for (int j = 0; j < 8; ++j) {
    int row = wave * 16 + j + hh * 8;
    float m = -1e30f;
#pragma unroll
    for (int t = 0; t < 4; ++t) {
      int col = t * 16 + l16;
      int dz = (row >> 4) - (col >> 4) + 3;
      int dy = ((row >> 2) & 3) - ((col >> 2) & 3) + 3;
      int dx = (row & 3) - (col & 3) + 3;
      float lg = acc[t][j] * scale + sm.rpbS[(dz * 49 + dy * 7 + dx) * NH + head];
      if (sm.rcS[row] != sm.rcS[col]) lg -= 100.f;
      acc[t][j] = lg;
      m = fmaxf(m, lg);
    }
#pragma unroll
    for (int off = 8; off > 0; off >>= 1) m = fmaxf(m, __shfl_xor(m, off, 32));
    float ssum = 0.f;
#pragma unroll
    for (int t = 0; t < 4; ++t) { float e = __expf(acc[t][j] - m); acc[t][j] = e; ssum += e; }
#pragma unroll
    for (int off = 8; off > 0; off >>= 1) ssum += __shfl_xor(ssum, off, 32);
    float inv = 1.f / ssum;
#pragma unroll
    for (int t = 0; t < 4; ++t) sm.pmat[row][t * 16 + l16] = (bf16_t)(acc[t][j] * inv);
  }
  __syncthreads();

  // o = P(64x64) @ V(64x32): wave -> M tile = wave, both N tiles
  v8f oc0 = {0,0,0,0,0,0,0,0}, oc1 = {0,0,0,0,0,0,0,0};
#pragma unroll
  for (int k0 = 0; k0 < NTOK; k0 += 32) {
    v16bf a  = load_frag_A(&sm.pmat[0][0], NTOK, wave * 16, k0, lane);
    v16bf b0 = load_frag_B(&sm.vT[0][0], NTOK, 0, k0, lane);
    v16bf b1 = load_frag_B(&sm.vT[0][0], NTOK, 16, k0, lane);
    oc0 = wmma_bf16(a, b0, oc0);
    oc1 = wmma_bf16(a, b1, oc1);
  }
  bf16_t* orow = obuf + (size_t)win * NTOK * C_DIM + head * HD;
#pragma unroll
  for (int j = 0; j < 8; ++j) {
    int tok = wave * 16 + j + hh * 8;
    orow[(size_t)tok * C_DIM + l16]      = (bf16_t)oc0[j];
    orow[(size_t)tok * C_DIM + 16 + l16] = (bf16_t)oc1[j];
  }
}

// ------------------------------------------ proj GEMM + unshift + residual
// grid(2048, 3), 128 threads, 2x2 tiles per wave. xout += win_rev(roll(A @ W + bias))
__global__ __launch_bounds__(128) void gemm_proj_add(const bf16_t* __restrict__ A,
                                                     const bf16_t* __restrict__ WT,
                                                     const float* __restrict__ bias,
                                                     float* __restrict__ xout, int shifted) {
  int wave = threadIdx.x >> 5, lane = threadIdx.x & 31;
  int win = blockIdx.x;
  int nBase = (wave >> 1) * 32;                   // token row within window
  int col0 = blockIdx.y * 64 + (wave & 1) * 32;
  v8f acc00 = {0,0,0,0,0,0,0,0}, acc01 = {0,0,0,0,0,0,0,0};
  v8f acc10 = {0,0,0,0,0,0,0,0}, acc11 = {0,0,0,0,0,0,0,0};
#pragma unroll
  for (int k0 = 0; k0 < C_DIM; k0 += 32) {
    v16bf a0 = load_frag_A(A, C_DIM, win * 64 + nBase,      k0, lane);
    v16bf a1 = load_frag_A(A, C_DIM, win * 64 + nBase + 16, k0, lane);
    v16bf b0 = load_frag_B(WT, C_DIM, col0,      k0, lane);
    v16bf b1 = load_frag_B(WT, C_DIM, col0 + 16, k0, lane);
    acc00 = wmma_bf16(a0, b0, acc00);
    acc01 = wmma_bf16(a0, b1, acc01);
    acc10 = wmma_bf16(a1, b0, acc10);
    acc11 = wmma_bf16(a1, b1, acc11);
  }
  int l16 = lane & 15, hh = lane >> 4;
  float bb0 = bias[col0 + l16], bb1 = bias[col0 + 16 + l16];
  int s4 = win >> 8, h4 = (win >> 4) & 15, w4 = win & 15;
#pragma unroll
  for (int j = 0; j < 8; ++j) {
#pragma unroll
    for (int i = 0; i < 2; ++i) {
      int n = nBase + i * 16 + j + hh * 8;
      int scc = s4 * 4 + (n >> 4), hcc = h4 * 4 + ((n >> 2) & 3), wcc = w4 * 4 + (n & 3);
      if (shifted) { scc = (scc + 2) & (S_DIM - 1); hcc = (hcc + 2) & (H_DIM - 1); wcc = (wcc + 2) & (W_DIM - 1); }
      size_t t = ((size_t)scc * H_DIM + hcc) * W_DIM + wcc;
      float a = (i == 0) ? acc00[j] : acc10[j];
      float c = (i == 0) ? acc01[j] : acc11[j];
      xout[t * C_DIM + col0 + l16]      += a + bb0;
      xout[t * C_DIM + col0 + 16 + l16] += c + bb1;
    }
  }
}

// ----------------------------------------------------------- fused MLP
// grid(2048), 256 threads. x += gelu(ln(x) @ f1w + f1b) @ f2w + f2b
__global__ __launch_bounds__(256) void mlp_fused(float* __restrict__ x,
                                                 const float* __restrict__ g,
                                                 const float* __restrict__ b,
                                                 const bf16_t* __restrict__ f1wT,
                                                 const float* __restrict__ f1b,
                                                 const bf16_t* __restrict__ f2wT,
                                                 const float* __restrict__ f2b) {
  __shared__ __align__(16) bf16_t xh[64][C_DIM];
  __shared__ __align__(16) bf16_t hb[64][64];
  int tid = threadIdx.x, wave = tid >> 5, lane = tid & 31;
  int l16 = lane & 15, hh = lane >> 4;
  int rowBlock = blockIdx.x * 64;

  for (int r8 = 0; r8 < 8; ++r8) {
    int row = wave * 8 + r8;
    const float* xr = x + (size_t)(rowBlock + row) * C_DIM;
    float v[6], s = 0.f;
#pragma unroll
    for (int i = 0; i < 6; ++i) { v[i] = xr[lane + 32 * i]; s += v[i]; }
#pragma unroll
    for (int off = 16; off > 0; off >>= 1) s += __shfl_xor(s, off, 32);
    float mean = s * (1.f / C_DIM), var = 0.f;
#pragma unroll
    for (int i = 0; i < 6; ++i) { float d = v[i] - mean; var += d * d; }
#pragma unroll
    for (int off = 16; off > 0; off >>= 1) var += __shfl_xor(var, off, 32);
    float rstd = rsqrtf(var * (1.f / C_DIM) + 1e-5f);
#pragma unroll
    for (int i = 0; i < 6; ++i) {
      int c = lane + 32 * i;
      xh[row][c] = (bf16_t)((v[i] - mean) * rstd * g[c] + b[c]);
    }
  }
  __syncthreads();

  int mT = wave >> 1;
  int colBase = (wave & 1) * 96;
  v8f oacc[6];
#pragma unroll
  for (int t = 0; t < 6; ++t) oacc[t] = (v8f){0,0,0,0,0,0,0,0};

  for (int chunk = 0; chunk < 12; ++chunk) {
    int hcol0 = chunk * 64 + (wave & 1) * 32;
    v8f h0 = {0,0,0,0,0,0,0,0}, h1 = {0,0,0,0,0,0,0,0};
#pragma unroll
    for (int k0 = 0; k0 < C_DIM; k0 += 32) {
      v16bf a  = load_frag_A(&xh[0][0], C_DIM, mT * 16, k0, lane);
      v16bf b0 = load_frag_B(f1wT, C_DIM, hcol0, k0, lane);
      v16bf b1 = load_frag_B(f1wT, C_DIM, hcol0 + 16, k0, lane);
      h0 = wmma_bf16(a, b0, h0);
      h1 = wmma_bf16(a, b1, h1);
    }
    float bb0 = f1b[hcol0 + l16], bb1 = f1b[hcol0 + 16 + l16];
#pragma unroll
    for (int j = 0; j < 8; ++j) {
      int r = mT * 16 + j + hh * 8;
      float u0 = h0[j] + bb0, u1 = h1[j] + bb1;
      u0 = 0.5f * u0 * (1.f + erff(u0 * 0.70710678118654752f));
      u1 = 0.5f * u1 * (1.f + erff(u1 * 0.70710678118654752f));
      hb[r][(wave & 1) * 32 + l16]      = (bf16_t)u0;
      hb[r][(wave & 1) * 32 + 16 + l16] = (bf16_t)u1;
    }
    __syncthreads();
#pragma unroll
    for (int kk = 0; kk < 64; kk += 32) {
      v16bf a = load_frag_A(&hb[0][0], 64, mT * 16, kk, lane);
#pragma unroll
      for (int t = 0; t < 6; ++t) {
        v16bf bf = load_frag_B(f2wT, C4, colBase + t * 16, chunk * 64 + kk, lane);
        oacc[t] = wmma_bf16(a, bf, oacc[t]);
      }
    }
    __syncthreads();
  }
#pragma unroll
  for (int t = 0; t < 6; ++t) {
    float bb = f2b[colBase + t * 16 + l16];
#pragma unroll
    for (int j = 0; j < 8; ++j) {
      int row = rowBlock + mT * 16 + j + hh * 8;
      size_t idx = (size_t)row * C_DIM + colBase + t * 16 + l16;
      x[idx] += oacc[t][j] + bb;
    }
  }
}

// ---------------------------------------------------------------- launch

extern "C" void kernel_launch(void* const* d_in, const int* in_sizes, int n_in,
                              void* d_out, int out_size, void* d_ws, size_t ws_size,
                              hipStream_t stream) {
  (void)in_sizes; (void)n_in; (void)out_size; (void)ws_size;
  const float* xin = (const float*)d_in[0];
  float* xo = (float*)d_out;

  bf16_t* wsb = (bf16_t*)d_ws;
  bf16_t* qwT[2]; bf16_t* pwT[2]; bf16_t* f1T[2]; bf16_t* f2T[2];
  size_t o = 0;
  for (int blk = 0; blk < 2; ++blk) {
    qwT[blk] = wsb + o; o += (size_t)C3 * C_DIM;
    pwT[blk] = wsb + o; o += (size_t)C_DIM * C_DIM;
    f1T[blk] = wsb + o; o += (size_t)C4 * C_DIM;
    f2T[blk] = wsb + o; o += (size_t)C_DIM * C4;
  }
  bf16_t* hbuf = wsb + o; o += (size_t)L_TOK * C_DIM;  // LN output / attention output (reused)
  bf16_t* qkvb = wsb + o;                              // qkv buffer

  copy_f32<<<4096, 256, 0, stream>>>(xin, xo, L_TOK * C_DIM / 4);

  for (int blk = 0; blk < 2; ++blk) {
    int base = 1 + 13 * blk;
    cvt_transpose<<<256, 256, 0, stream>>>((const float*)d_in[base + 2],  qwT[blk], C_DIM, C3);
    cvt_transpose<<<256, 256, 0, stream>>>((const float*)d_in[base + 5],  pwT[blk], C_DIM, C_DIM);
    cvt_transpose<<<256, 256, 0, stream>>>((const float*)d_in[base + 9],  f1T[blk], C_DIM, C4);
    cvt_transpose<<<256, 256, 0, stream>>>((const float*)d_in[base + 11], f2T[blk], C4, C_DIM);
  }

  for (int blk = 0; blk < 2; ++blk) {
    int base = 1 + 13 * blk;
    const float* g1  = (const float*)d_in[base + 0];
    const float* b1  = (const float*)d_in[base + 1];
    const float* qb  = (const float*)d_in[base + 3];
    const float* rpb = (const float*)d_in[base + 4];
    const float* pb  = (const float*)d_in[base + 6];
    const float* g2  = (const float*)d_in[base + 7];
    const float* b2  = (const float*)d_in[base + 8];
    const float* f1b = (const float*)d_in[base + 10];
    const float* f2b = (const float*)d_in[base + 12];
    int shifted = blk;

    ln_part<<<L_TOK / 8, 256, 0, stream>>>(xo, g1, b1, hbuf, shifted);
    gemm_qkv<<<dim3(NWIN, C3 / 64), 128, 0, stream>>>(hbuf, qwT[blk], qb, qkvb);
    attn_kernel<<<dim3(NH, NWIN), 128, 0, stream>>>(qkvb, rpb, hbuf, shifted);
    gemm_proj_add<<<dim3(NWIN, C_DIM / 64), 128, 0, stream>>>(hbuf, pwT[blk], pb, xo, shifted);
    mlp_fused<<<L_TOK / 64, 256, 0, stream>>>(xo, g2, b2, f1T[blk], f1b, f2T[blk], f2b);
  }
}
